// BatchedTorchParametricSolver_81767587381598
// MI455X (gfx1250) — matrix operand order repaired and checked by reading, not compiled, not run on hardware
//
#include <hip/hip_runtime.h>
#include <hip/hip_bf16.h>

#define N_ELEM_C 196608   // 3 * 256 * 256
#define N_OPS_C  65536    // 256 * 256
#define MEM_BLKS 768      // 196608 / 256
#define OP_BLKS  256      // 65536 / 256
#define N_ROWS_C 24576    // N_ELEM / 8

typedef __attribute__((ext_vector_type(16))) __bf16 v16bf;
typedef __attribute__((ext_vector_type(8)))  float  v8f;

union BF16x16 { unsigned short u[16]; v16bf v; };

__device__ __forceinline__ unsigned flipf(float f) {
  unsigned u = __float_as_uint(f);
  return u ^ ((u >> 31) ? 0xFFFFFFFFu : 0x80000000u);   // ascending-order map
}
__device__ __forceinline__ unsigned short bf16t(float f) {
  return (unsigned short)(__float_as_uint(f) >> 16);     // truncate to bf16
}

// ---------------------------------------------------------------- sort keys
__global__ void k_init_mem_keys(const float* __restrict__ logits,
                                const float* __restrict__ gum,
                                unsigned* __restrict__ keys,
                                unsigned* __restrict__ idx, int n) {
  int b = blockIdx.y;
  int i = blockIdx.x * 256 + threadIdx.x;
  float v = logits[(size_t)b * n + i] + gum[(size_t)b * n + i];
  keys[(size_t)b * n + i] = flipf(v);
  idx[(size_t)b * n + i]  = (unsigned)i;
}

// ---------------------------------------------------------------- radix sort
__global__ void k_radix_hist(const unsigned* __restrict__ keys,
                             unsigned* __restrict__ hist,
                             int n, int nblocks, int shift) {
  __shared__ unsigned cnt[256];
  int t = threadIdx.x, b = blockIdx.y;
  cnt[t] = 0; __syncthreads();
  unsigned d = (keys[(size_t)b * n + blockIdx.x * 256 + t] >> shift) & 255u;
  atomicAdd(&cnt[d], 1u);
  __syncthreads();
  hist[(size_t)b * 256 * nblocks + (size_t)t * nblocks + blockIdx.x] = cnt[t];
}

__global__ void k_radix_scan(unsigned* __restrict__ hist,
                             unsigned* __restrict__ digitBase, int nblocks) {
  int b = blockIdx.x, d = threadIdx.x;
  size_t base = (size_t)b * 256 * nblocks + (size_t)d * nblocks;
  unsigned run = 0;
  for (int k = 0; k < nblocks; k++) { unsigned v = hist[base + k]; hist[base + k] = run; run += v; }
  __shared__ unsigned tot[256];
  tot[d] = run; __syncthreads();
  if (d == 0) { unsigned acc = 0; for (int k = 0; k < 256; k++) { unsigned v = tot[k]; tot[k] = acc; acc += v; } }
  __syncthreads();
  digitBase[b * 256 + d] = tot[d];
}

__global__ void k_radix_scatter(const unsigned* __restrict__ keysIn,
                                const unsigned* __restrict__ idxIn,
                                unsigned* __restrict__ keysOut,
                                unsigned* __restrict__ idxOut,
                                const unsigned* __restrict__ hist,
                                const unsigned* __restrict__ digitBase,
                                int n, int nblocks, int shift) {
  __shared__ unsigned sdig[256];
  int t = threadIdx.x, b = blockIdx.y;
  size_t gi = (size_t)b * n + blockIdx.x * 256 + t;
  unsigned key = keysIn[gi], idv = idxIn[gi];
  unsigned d = (key >> shift) & 255u;
  sdig[t] = d; __syncthreads();
  unsigned rank = 0;
  for (int j = 0; j < t; j++) rank += (sdig[j] == d);           // stable in-block rank
  unsigned pos = digitBase[b * 256 + d]
               + hist[(size_t)b * 256 * nblocks + (size_t)d * nblocks + blockIdx.x] + rank;
  keysOut[(size_t)b * n + pos] = key;
  idxOut [(size_t)b * n + pos] = idv;
}

// ---------------------------------------------------- Plackett-Luce logprob
__global__ void k_pl_max(const float* __restrict__ logits, const unsigned* __restrict__ order,
                         int n, float* __restrict__ pmax) {
  __shared__ float red[256];
  int b = blockIdx.y, t = threadIdx.x, i = blockIdx.x * 256 + t;
  red[t] = logits[(size_t)b * n + order[(size_t)b * n + i]];
  __syncthreads();
  for (int st = 128; st > 0; st >>= 1) { if (t < st) red[t] = fmaxf(red[t], red[t + st]); __syncthreads(); }
  if (t == 0) pmax[b * gridDim.x + blockIdx.x] = red[0];
}
__global__ void k_pl_maxfin(const float* __restrict__ pmax, int nblocks, float* __restrict__ Mv) {
  __shared__ float red[256];
  int b = blockIdx.x, t = threadIdx.x;
  float m = -3.4e38f;
  for (int k = t; k < nblocks; k += 256) m = fmaxf(m, pmax[b * nblocks + k]);
  red[t] = m; __syncthreads();
  for (int st = 128; st > 0; st >>= 1) { if (t < st) red[t] = fmaxf(red[t], red[t + st]); __syncthreads(); }
  if (t == 0) Mv[b] = red[0];
}
__global__ void k_pl_exp(const float* __restrict__ logits, const unsigned* __restrict__ order,
                         int n, const float* __restrict__ Mv,
                         float* __restrict__ eblk, float* __restrict__ sblk) {
  __shared__ float re[256], rs[256];
  int b = blockIdx.y, t = threadIdx.x, i = blockIdx.x * 256 + t;
  float s = logits[(size_t)b * n + order[(size_t)b * n + i]];
  re[t] = expf(s - Mv[b]); rs[t] = s; __syncthreads();
  for (int st = 128; st > 0; st >>= 1) { if (t < st) { re[t] += re[t + st]; rs[t] += rs[t + st]; } __syncthreads(); }
  if (t == 0) { eblk[b * gridDim.x + blockIdx.x] = re[0]; sblk[b * gridDim.x + blockIdx.x] = rs[0]; }
}
__global__ void k_pl_suffix(const float* __restrict__ eblk, const float* __restrict__ sblk,
                            int nblocks, float* __restrict__ suf, float* __restrict__ sumS) {
  int b = blockIdx.x;
  if (threadIdx.x == 0) {
    float run = 0;
    for (int k = nblocks - 1; k >= 0; k--) { suf[b * nblocks + k] = run; run += eblk[b * nblocks + k]; }
    float ss = 0;
    for (int k = 0; k < nblocks; k++) ss += sblk[b * nblocks + k];
    sumS[b] = ss;
  }
}
__global__ void k_pl_log(const float* __restrict__ logits, const unsigned* __restrict__ order,
                         int n, const float* __restrict__ Mv, const float* __restrict__ suf,
                         float* __restrict__ plog) {
  __shared__ float e[256], red[256];
  int b = blockIdx.y, t = threadIdx.x, i = blockIdx.x * 256 + t;
  float s = logits[(size_t)b * n + order[(size_t)b * n + i]];
  e[t] = expf(s - Mv[b]); __syncthreads();
  if (t == 0) { float run = 0; for (int k = 255; k >= 0; k--) { run += e[k]; e[k] = run; } }
  __syncthreads();
  float T = e[t] + suf[b * gridDim.x + blockIdx.x];
  red[t] = logf(T); __syncthreads();
  for (int st = 128; st > 0; st >>= 1) { if (t < st) red[t] += red[t + st]; __syncthreads(); }
  if (t == 0) plog[b * gridDim.x + blockIdx.x] = red[0];
}
__global__ void k_pl_final(const float* __restrict__ plog, const float* __restrict__ sumS,
                           const float* __restrict__ Mv, int n, int nblocks,
                           float* __restrict__ out, int row) {
  int b = blockIdx.x;
  if (threadIdx.x == 0) {
    float sl = 0;
    for (int k = 0; k < nblocks; k++) sl += plog[b * nblocks + k];
    out[row * 8 + b] = sumS[b] - sl - (float)n * Mv[b];
  }
}

// ---------------------------------------------------------- per-matrix conv
__global__ void k_pm_conv(const unsigned* __restrict__ perm,
                          const float* __restrict__ pw, const float* __restrict__ pb,
                          float* __restrict__ mem_flat) {
  __shared__ float w[72]; __shared__ float bias[8];
  int t = threadIdx.x, m = blockIdx.y, b = blockIdx.z;
  if (t < 72) w[t] = pw[m * 72 + t];
  if (t < 8)  bias[t] = pb[m * 8 + t];
  __syncthreads();
  int p = blockIdx.x * 256 + t, i = p >> 8, j = p & 255;
  const unsigned* ps = perm + (size_t)b * N_ELEM_C + (size_t)m * N_OPS_C;
  float acc[8];
  for (int oc = 0; oc < 8; oc++) acc[oc] = bias[oc];
  for (int ky = 0; ky < 3; ky++) {
    int ii = i + ky - 1; if ((unsigned)ii > 255u) continue;
    for (int kx = 0; kx < 3; kx++) {
      int jj = j + kx - 1; if ((unsigned)jj > 255u) continue;
      float val = (float)ps[ii * 256 + jj];
      for (int oc = 0; oc < 8; oc++) acc[oc] += w[oc * 9 + ky * 3 + kx] * val;
    }
  }
  unsigned a = ps[p];                       // scatter address (perm value)
  for (int oc = 0; oc < 8; oc++) {
    float v = acc[oc]; v = v > 0.f ? v : 0.f;
    mem_flat[(size_t)b * 8 * N_ELEM_C + (size_t)oc * N_ELEM_C + a] = v;
  }
}

// --------------------------------------------- mem conv (8->16) + pool part
__global__ void k_mc_conv(const float* __restrict__ mem_flat,
                          const float* __restrict__ mw, const float* __restrict__ mb,
                          float* __restrict__ partial) {
  __shared__ float w[1152]; __shared__ float bias[16]; __shared__ float vals[256 * 16];
  int t = threadIdx.x, b = blockIdx.y;
  for (int k = t; k < 1152; k += 256) w[k] = mw[k];
  if (t < 16) bias[t] = mb[t];
  __syncthreads();
  int p = blockIdx.x * 256 + t, row = p >> 3, col = p & 7;
  const float* mf = mem_flat + (size_t)b * 8 * N_ELEM_C;
  float acc[16];
  for (int oc = 0; oc < 16; oc++) acc[oc] = bias[oc];
  for (int ic = 0; ic < 8; ic++) {
    const float* ch = mf + (size_t)ic * N_ELEM_C;
    for (int ky = 0; ky < 3; ky++) {
      int r = row + ky - 1; if ((unsigned)r >= (unsigned)N_ROWS_C) continue;
      for (int kx = 0; kx < 3; kx++) {
        int c = col + kx - 1; if ((unsigned)c > 7u) continue;
        float val = ch[(size_t)r * 8 + c];
        for (int oc = 0; oc < 16; oc++) acc[oc] += w[oc * 72 + ic * 9 + ky * 3 + kx] * val;
      }
    }
  }
  for (int oc = 0; oc < 16; oc++) { float v = acc[oc]; vals[t * 16 + oc] = v > 0.f ? v : 0.f; }
  __syncthreads();
  if (t < 64) {                             // t = oc*4 + cg ; fixed-order sum
    int oc = t >> 2, cg = t & 3;
    float s = 0;
    for (int rl = 0; rl < 32; rl++) {
      s += vals[(rl * 8 + cg * 2) * 16 + oc];
      s += vals[(rl * 8 + cg * 2 + 1) * 16 + oc];
    }
    partial[((size_t)b * MEM_BLKS + blockIdx.x) * 64 + t] = s;
  }
}

__global__ void k_pool_reduce(const float* __restrict__ partial, float* __restrict__ pooled) {
  int b = blockIdx.x, t = threadIdx.x;       // t = oc*16 + rg*4 + cg (matches JAX flatten)
  int oc = t >> 4, rg = (t >> 2) & 3, cg = t & 3;
  float s = 0;
  for (int k = 0; k < 192; k++) {
    int blk = rg * 192 + k;
    s += partial[((size_t)b * MEM_BLKS + blk) * 64 + oc * 4 + cg];
  }
  pooled[b * 256 + t] = s * (1.0f / 12288.0f);
}

// -------------------------------------------- projection GEMM: WMMA bf16
// D(65536x8) = W(65536x256) * P(256x8) ; one wave per 16-row tile of W.
// P is staged into LDS via async global->LDS copy; a zeroed 9th region lets
// N>=8 lanes build zero B columns with no exec-mask divergence.
__global__ void k_proj_wmma(const float* __restrict__ projw, const float* __restrict__ projb,
                            const float* __restrict__ pooled, const float* __restrict__ gop,
                            float* __restrict__ op_logits,
                            unsigned* __restrict__ okeys, unsigned* __restrict__ oidx) {
  __shared__ float plds[9 * 256];
  int t = threadIdx.x;
  plds[8 * 256 + t] = 0.0f;                 // zero pad region for N>=8
  // async-stage pooled (8x256 f32, 8KB) into LDS: 8 elements per thread
  for (int j = 0; j < 8; j++) {
    unsigned elem = (unsigned)(j * 256 + t);
    unsigned lds_off = (unsigned)(size_t)&plds[elem];   // LDS byte offset (low 32b of shared addr)
    const float* ga = pooled + elem;
    asm volatile("global_load_async_to_lds_b32 %0, %1, off"
                 :: "v"(lds_off), "v"(ga) : "memory");
  }
  asm volatile("s_wait_asynccnt 0x0" ::: "memory");
  __syncthreads();

  int wave = t >> 5, lane = t & 31;
  int mt = blockIdx.x * 8 + wave;            // 0..4095 tiles of 16 rows
  int mrow = lane & 15, khalf = lane >> 4;
  int N = lane & 15;
  int nb = (N < 8) ? N : 8;                  // branch-free zero column select
  v8f c = {0.f, 0.f, 0.f, 0.f, 0.f, 0.f, 0.f, 0.f};
  for (int kc = 0; kc < 8; kc++) {
    int kb = kc * 32 + 16 * khalf;
    // B fragment from LDS: lane N, K = kb + [0..15]
    BF16x16 bb;
    for (int i2 = 0; i2 < 16; i2++) bb.u[i2] = bf16t(plds[nb * 256 + kb + i2]);
    // A fragment (contiguous k, same order as B): lane M=mrow, K = 16h + [0..15]
    BF16x16 aa;
    const float* rowp = projw + (size_t)(mt * 16 + mrow) * 256 + kc * 32;
    __builtin_prefetch(rowp + 32, 0, 1);     // next K-chunk -> global_prefetch_b8
    for (int i2 = 0; i2 < 8; i2++) aa.u[i2]     = bf16t(rowp[16 * khalf + i2]);
    for (int i2 = 0; i2 < 8; i2++) aa.u[8 + i2] = bf16t(rowp[16 * khalf + 8 + i2]);
    c = __builtin_amdgcn_wmma_f32_16x16x32_bf16(false, aa.v, false, bb.v,
                                                (short)0, c, false, false);
  }
  // C/D layout: lane N = lane&15 ; VGPR r -> M = r + 8*khalf
  for (int r = 0; r < 8; r++) {
    int g = mt * 16 + r + 8 * khalf;
    float logit = c[r] + projb[g];           // bias add hoisted (uniform in N)
    if (N < 8) {
      size_t o = (size_t)N * N_OPS_C + g;
      op_logits[o] = logit;
      okeys[o] = flipf(logit + gop[o]);
      oidx [o] = (unsigned)g;
    }
  }
}

// ----------------------------------------------------------- penalties
__device__ __forceinline__ float tierf(float h) {
  return h <= 2.f ? 1.f : h <= 4.f ? 1.5f : h <= 8.f ? 2.f : h <= 16.f ? 3.f : 5.f;
}
__device__ __forceinline__ float penf(float d) {
  float f = d > 0.f ? d : 0.f;
  float bk = d < 0.f ? -d : 0.f;
  return f * tierf(f) + bk * bk * tierf(bk);
}

__global__ void k_penalty(const unsigned* __restrict__ perm, const unsigned* __restrict__ oidx,
                          float* __restrict__ pinter, float* __restrict__ pintra) {
  __shared__ float ri[256], ra[256];
  int b = blockIdx.y, t = threadIdx.x, p = blockIdx.x * 256 + t;
  const unsigned* pm = perm + (size_t)b * N_ELEM_C;
  const unsigned* od = oidx + (size_t)b * N_OPS_C;
  unsigned o = od[p];
  float v0 = (float)pm[o], v1 = (float)pm[N_OPS_C + o], v2 = (float)pm[2 * N_OPS_C + o];
  float intra = penf(v1 - v0) + penf(v2 - v1);
  float inter = 0.f;
  if (p < N_OPS_C - 1) { unsigned on = od[p + 1]; inter = penf((float)pm[on] - v2); }
  ri[t] = inter; ra[t] = intra; __syncthreads();
  for (int st = 128; st > 0; st >>= 1) { if (t < st) { ri[t] += ri[t + st]; ra[t] += ra[t + st]; } __syncthreads(); }
  if (t == 0) { pinter[b * 256 + blockIdx.x] = ri[0]; pintra[b * 256 + blockIdx.x] = ra[0]; }
}
__global__ void k_penalty_final(const float* __restrict__ pinter, const float* __restrict__ pintra,
                                float* __restrict__ out) {
  int b = blockIdx.x;
  if (threadIdx.x == 0) {
    float si = 0, sa = 0;
    for (int k = 0; k < 256; k++) { si += pinter[b * 256 + k]; sa += pintra[b * 256 + k]; }
    out[0 * 8 + b] = si;
    out[1 * 8 + b] = sa;
  }
}

// ----------------------------------------------------------------- driver
static void radix_sort(unsigned* k0, unsigned* i0, unsigned* k1, unsigned* i1,
                       unsigned* hist, unsigned* digitBase,
                       int n, int nblocks, hipStream_t stream) {
  unsigned *ka = k0, *ia = i0, *kb = k1, *ib = i1;
  for (int pass = 0; pass < 4; pass++) {
    int shift = pass * 8;
    k_radix_hist<<<dim3(nblocks, 8), 256, 0, stream>>>(ka, hist, n, nblocks, shift);
    k_radix_scan<<<8, 256, 0, stream>>>(hist, digitBase, nblocks);
    k_radix_scatter<<<dim3(nblocks, 8), 256, 0, stream>>>(ka, ia, kb, ib, hist, digitBase, n, nblocks, shift);
    unsigned* tk = ka; ka = kb; kb = tk;
    unsigned* ti = ia; ia = ib; ib = ti;
  }
  // 4 passes -> result ends in (k0, i0)
}

static void pl_logprob(const float* logits, const unsigned* order, int n, int nblocks,
                       float* pmax, float* Mv, float* eblk, float* sblk, float* suf,
                       float* sumS, float* plog, float* out, int row, hipStream_t stream) {
  k_pl_max   <<<dim3(nblocks, 8), 256, 0, stream>>>(logits, order, n, pmax);
  k_pl_maxfin<<<8, 256, 0, stream>>>(pmax, nblocks, Mv);
  k_pl_exp   <<<dim3(nblocks, 8), 256, 0, stream>>>(logits, order, n, Mv, eblk, sblk);
  k_pl_suffix<<<8, 1, 0, stream>>>(eblk, sblk, nblocks, suf, sumS);
  k_pl_log   <<<dim3(nblocks, 8), 256, 0, stream>>>(logits, order, n, Mv, suf, plog);
  k_pl_final <<<8, 1, 0, stream>>>(plog, sumS, Mv, n, nblocks, out, row);
}

extern "C" void kernel_launch(void* const* d_in, const int* in_sizes, int n_in,
                              void* d_out, int out_size, void* d_ws, size_t ws_size,
                              hipStream_t stream) {
  const float* mem_logits = (const float*)d_in[0];
  const float* gumbel_mem = (const float*)d_in[1];
  const float* gumbel_op  = (const float*)d_in[2];
  const float* pm_w       = (const float*)d_in[3];
  const float* pm_b       = (const float*)d_in[4];
  const float* mc_w       = (const float*)d_in[5];
  const float* mc_b       = (const float*)d_in[6];
  const float* proj_w     = (const float*)d_in[7];
  const float* proj_b     = (const float*)d_in[8];
  float* out = (float*)d_out;

  char* ws = (char*)d_ws;
  size_t off = 0;
  auto alloc = [&](size_t bytes) -> void* {
    void* p = ws + off;
    off += (bytes + 255) & ~(size_t)255;
    return p;
  };

  unsigned* mk0 = (unsigned*)alloc((size_t)8 * N_ELEM_C * 4);
  unsigned* mk1 = (unsigned*)alloc((size_t)8 * N_ELEM_C * 4);
  unsigned* mi0 = (unsigned*)alloc((size_t)8 * N_ELEM_C * 4);
  unsigned* mi1 = (unsigned*)alloc((size_t)8 * N_ELEM_C * 4);
  unsigned* ok0 = (unsigned*)alloc((size_t)8 * N_OPS_C * 4);
  unsigned* ok1 = (unsigned*)alloc((size_t)8 * N_OPS_C * 4);
  unsigned* oi0 = (unsigned*)alloc((size_t)8 * N_OPS_C * 4);
  unsigned* oi1 = (unsigned*)alloc((size_t)8 * N_OPS_C * 4);
  unsigned* hist      = (unsigned*)alloc((size_t)8 * 256 * MEM_BLKS * 4);
  unsigned* digitBase = (unsigned*)alloc((size_t)8 * 256 * 4);
  float* mem_flat  = (float*)alloc((size_t)8 * 8 * N_ELEM_C * 4);
  float* mcPartial = (float*)alloc((size_t)8 * MEM_BLKS * 64 * 4);
  float* pooled    = (float*)alloc((size_t)8 * 256 * 4);
  float* op_logits = (float*)alloc((size_t)8 * N_OPS_C * 4);
  float* pmax = (float*)alloc((size_t)8 * MEM_BLKS * 4);
  float* eblk = (float*)alloc((size_t)8 * MEM_BLKS * 4);
  float* sblk = (float*)alloc((size_t)8 * MEM_BLKS * 4);
  float* suf  = (float*)alloc((size_t)8 * MEM_BLKS * 4);
  float* plog = (float*)alloc((size_t)8 * MEM_BLKS * 4);
  float* Mv   = (float*)alloc(8 * 4);
  float* sumS = (float*)alloc(8 * 4);
  float* pinter = (float*)alloc((size_t)8 * 256 * 4);
  float* pintra = (float*)alloc((size_t)8 * 256 * 4);

  // 1) memory permutation: keys + stable radix argsort (perm -> mi0)
  k_init_mem_keys<<<dim3(MEM_BLKS, 8), 256, 0, stream>>>(mem_logits, gumbel_mem, mk0, mi0, N_ELEM_C);
  radix_sort(mk0, mi0, mk1, mi1, hist, digitBase, N_ELEM_C, MEM_BLKS, stream);

  // 2) mem Plackett-Luce logprob -> out row 3
  pl_logprob(mem_logits, mi0, N_ELEM_C, MEM_BLKS, pmax, Mv, eblk, sblk, suf, sumS, plog,
             out, 3, stream);

  // 3) per-matrix convs + scatter into mem_flat (every column written once)
  k_pm_conv<<<dim3(256, 3, 8), 256, 0, stream>>>(mi0, pm_w, pm_b, mem_flat);

  // 4) mem conv (8->16) + deterministic two-stage pooling
  k_mc_conv<<<dim3(MEM_BLKS, 8), 256, 0, stream>>>(mem_flat, mc_w, mc_b, mcPartial);
  k_pool_reduce<<<8, 256, 0, stream>>>(mcPartial, pooled);

  // 5) projection GEMM via v_wmma_f32_16x16x32_bf16 (async-LDS staged B),
  //    fused epilogue builds op sort keys
  k_proj_wmma<<<512, 256, 0, stream>>>(proj_w, proj_b, pooled, gumbel_op, op_logits, ok0, oi0);

  // 6) op argsort (order -> oi0)
  radix_sort(ok0, oi0, ok1, oi1, hist, digitBase, N_OPS_C, OP_BLKS, stream);

  // 7) op Plackett-Luce logprob -> out row 2
  pl_logprob(op_logits, oi0, N_OPS_C, OP_BLKS, pmax, Mv, eblk, sblk, suf, sumS, plog,
             out, 2, stream);

  // 8) inter/intra penalties -> out rows 0,1
  k_penalty<<<dim3(OP_BLKS, 8), 256, 0, stream>>>(mi0, oi0, pinter, pintra);
  k_penalty_final<<<8, 1, 0, stream>>>(pinter, pintra, out);
}
